// MultiHeadAttention_24318104830456
// MI455X (gfx1250) — matrix operand order repaired and checked by reading, not compiled, not run on hardware
//
#include <hip/hip_runtime.h>
#include <hip/hip_bf16.h>

typedef __attribute__((ext_vector_type(16))) _Float16     v16h;
typedef __attribute__((ext_vector_type(8)))  _Float16     v8h;
typedef __attribute__((ext_vector_type(8)))  float        v8f;
typedef __attribute__((ext_vector_type(4)))  int          v4i;
typedef __attribute__((ext_vector_type(8)))  int          v8i;
typedef __attribute__((ext_vector_type(4)))  unsigned int v4u;

#define HID 512
#define BM 128
#define BN 128
#define BK 32

// ---- CDNA5 async global->LDS copy (ASYNCcnt-tracked), with safe fallback ----
#if defined(__has_builtin)
#  if __has_builtin(__builtin_amdgcn_global_load_async_to_lds_b128)
#    define ASYNC_CP_OK 1
#  endif
#  if __has_builtin(__builtin_amdgcn_s_wait_asynccnt)
#    define WAITASYNC(n) __builtin_amdgcn_s_wait_asynccnt(n)
#  endif
#  if __has_builtin(__builtin_amdgcn_tensor_load_to_lds) && \
      __has_builtin(__builtin_amdgcn_s_wait_tensorcnt)
#    define TDM_OK 1
#  endif
#endif
#ifndef ASYNC_CP_OK
#  define ASYNC_CP_OK 0
#endif
#ifndef WAITASYNC
#  define WAITASYNC(n) asm volatile("s_wait_asynccnt %0" ::"n"(n) : "memory")
#endif
#ifndef TDM_OK
#  define TDM_OK 0
#endif

static __device__ inline void async_cp16(const _Float16* g, _Float16* l) {
#if ASYNC_CP_OK
    __builtin_amdgcn_global_load_async_to_lds_b128(
        (__attribute__((address_space(1))) v4i*)(const void*)g,
        (__attribute__((address_space(3))) v4i*)(void*)l,
        /*offset=*/0, /*cpol=*/0);
#else
    *(v8h*)l = *(const v8h*)g;
#endif
}

#if TDM_OK
// Tensor Data Mover: 2D f16 tile (tile_w x tile_h) global->LDS, row pitch in
// elements. Descriptor per CDNA5 ISA ch.8 (group0 + group1, groups 2/3 NULL).
static __device__ inline void tdm_load_tile_2d(const void* gsrc, unsigned lds_off,
                                               unsigned tile_w, unsigned tile_h,
                                               unsigned pitch_elems) {
    const unsigned long long ga = (unsigned long long)(size_t)gsrc;
    v4u g0;
    g0[0] = 1u;                                  // count=1 valid descriptor
    g0[1] = lds_off;                             // lds_addr (bytes)
    g0[2] = (unsigned)ga;                        // global_addr[31:0]
    g0[3] = (unsigned)(ga >> 32) | (2u << 30);   // global_addr[56:32] | type=2
    v8i g1;
    g1[0] = (int)(1u << 16);                     // data_size=1 (2B); no multicast
    g1[1] = (int)((tile_w & 0xFFFFu) << 16);     // tensor_dim0[15:0]
    g1[2] = (int)((tile_w >> 16) | ((tile_h & 0xFFFFu) << 16));  // dim0 hi | dim1 lo
    g1[3] = (int)((tile_h >> 16) | ((tile_w & 0xFFFFu) << 16));  // dim1 hi | tile_dim0
    g1[4] = (int)(tile_h & 0xFFFFu);             // tile_dim1 (tile_dim2=0)
    g1[5] = (int)pitch_elems;                    // tensor_dim0_stride[31:0]
    g1[6] = 0;
    g1[7] = 0;
    v4i z4 = {0, 0, 0, 0};
#  if __clang_major__ >= 23
    v8i z8 = {0, 0, 0, 0, 0, 0, 0, 0};
    __builtin_amdgcn_tensor_load_to_lds(g0, g1, z4, z4, z8, 0);
#  else
    __builtin_amdgcn_tensor_load_to_lds(g0, g1, z4, z4, 0);
#  endif
}
#endif

static __device__ inline v16h cat8(v8h lo, v8h hi) {
    union { v16h v; v8h h[2]; } u;
    u.h[0] = lo; u.h[1] = hi;
    return u.v;
}

static __device__ inline v8f wmma_f16(v16h a, v16h b, v8f c) {
    // (neg_a, A, neg_b, B, c_mod, C, reuse_a, reuse_b)
    return __builtin_amdgcn_wmma_f32_16x16x32_f16(false, a, false, b, (short)0, c,
                                                  false, false);
}

// ---------------------------------------------------------------------------
// Elementwise f32 -> f16 cast (n multiple of 2048)
// ---------------------------------------------------------------------------
__global__ __launch_bounds__(256)
void cast_f16(const float* __restrict__ src, _Float16* __restrict__ dst, int n) {
    const int i = (blockIdx.x * 256 + threadIdx.x) * 8;
    if (i >= n) return;
    const float4 a = *(const float4*)(src + i);
    const float4 b = *(const float4*)(src + i + 4);
    v8h o;
    o[0] = (_Float16)a.x; o[1] = (_Float16)a.y;
    o[2] = (_Float16)a.z; o[3] = (_Float16)a.w;
    o[4] = (_Float16)b.x; o[5] = (_Float16)b.y;
    o[6] = (_Float16)b.z; o[7] = (_Float16)b.w;
    *(v8h*)(dst + i) = o;
}

// ---------------------------------------------------------------------------
// C[M,HID] = A[M,HID] @ W[HID,HID]^T + bias (A, W f16; acc f32).
// Double-buffered async-to-LDS staging: 4 b128 async ops / wave / K-tile.
// ---------------------------------------------------------------------------
template <bool OUT_F16>
__global__ __launch_bounds__(256)
void gemm_xwT(const _Float16* __restrict__ A, const _Float16* __restrict__ W,
              const float* __restrict__ bias, void* __restrict__ outp,
              int M, float outScale) {
    __shared__ __align__(32) _Float16 As[2][BM * BK];
    __shared__ __align__(32) _Float16 Bs[2][BN * BK];

    const int tid  = threadIdx.x;
    const int wave = tid >> 5;
    const int lane = tid & 31;
    const int hl   = lane >> 4;
    const int l16  = lane & 15;
    const int wm   = wave & 3;    // 4 row-slices of 32
    const int wn   = wave >> 2;   // 2 col-slices of 64
    const int m0   = blockIdx.x * BM;
    const int n0   = blockIdx.y * BN;

    v8f acc[2][4] = {};

    // stage one 128x32 + 128x32 f16 tile pair into buffer `buf`
    auto stage = [&](int k0, int buf) {
#pragma unroll
        for (int p = 0; p < 2; ++p) {
            const int o    = tid * 16 + p * 4096;  // byte offset in 8KB tile
            const int row  = o >> 6;               // 64B (32 halves) per row
            const int colh = (o & 63) >> 1;
            async_cp16(A + (size_t)(m0 + row) * HID + k0 + colh,
                       (_Float16*)((char*)As[buf] + o));
            async_cp16(W + (size_t)(n0 + row) * HID + k0 + colh,
                       (_Float16*)((char*)Bs[buf] + o));
        }
    };

    stage(0, 0);
    int buf = 0;
    for (int k0 = 0; k0 < HID; k0 += BK, buf ^= 1) {
        const bool more = (k0 + BK) < HID;
        if (more) {
            stage(k0 + BK, buf ^ 1);
            WAITASYNC(4);   // current tile's 4 ops retired; next 4 in flight
        } else {
            WAITASYNC(0);
        }
        __syncthreads();

#pragma unroll
        for (int mt = 0; mt < 2; ++mt) {
            const _Float16* arow = &As[buf][(wm * 32 + mt * 16 + l16) * BK];
            v16h a = *(const v16h*)(arow + hl * 16);
#pragma unroll
            for (int nt = 0; nt < 4; ++nt) {
                v16h b = *(const v16h*)&Bs[buf][(wn * 64 + nt * 16 + l16) * BK + hl * 16];
                acc[mt][nt] = wmma_f16(a, b, acc[mt][nt]);
            }
        }
        __syncthreads();
    }

    // epilogue: C layout -> lane l16 = col, vgpr rr (+8*hl) = row
#pragma unroll
    for (int mt = 0; mt < 2; ++mt) {
#pragma unroll
        for (int nt = 0; nt < 4; ++nt) {
            const int n    = n0 + wn * 64 + nt * 16 + l16;
            const float bv = bias[n];
#pragma unroll
            for (int rr = 0; rr < 8; ++rr) {
                const int m   = m0 + wm * 32 + mt * 16 + rr + hl * 8;
                const float v = (acc[mt][nt][rr] + bv) * outScale;
                if (OUT_F16)
                    ((_Float16*)outp)[(size_t)m * HID + n] = (_Float16)v;
                else
                    ((float*)outp)[(size_t)m * HID + n] = v;
            }
        }
    }
}

// ---------------------------------------------------------------------------
// Flash attention: one block per (b, h, 128-row Q block). Q pre-scaled 1/8.
// Q async-to-LDS; K tile via Tensor Data Mover; V staged transposed manually.
// ---------------------------------------------------------------------------
__global__ __launch_bounds__(256)
void flash_attn(const _Float16* __restrict__ Qh, const _Float16* __restrict__ Kh,
                const _Float16* __restrict__ Vh, _Float16* __restrict__ Ctx) {
    const int S = 2048, D = 64;
    const int idx    = blockIdx.x;
    const int mt_blk = idx & 15;        // S/128 = 16
    const int h      = (idx >> 4) & 7;  // 8 heads
    const int b      = idx >> 7;        // batch
    const size_t base = (size_t)b * S * HID + (size_t)h * D;

    __shared__ __align__(32) _Float16 Qs[128 * 64];
    __shared__ __align__(32) _Float16 Ks[64 * 64];
    __shared__ __align__(32) _Float16 Vts[64 * 64];  // transposed: [d][s]
    __shared__ __align__(32) _Float16 Ps[128 * 64];

    const int tid  = threadIdx.x;
    const int wave = tid >> 5;
    const int lane = tid & 31;
    const int hl   = lane >> 4;
    const int l16  = lane & 15;

    // ---- async-stage the 128x64 Q tile (16KB -> 4 b128 ops/wave) ----
#pragma unroll
    for (int p = 0; p < 4; ++p) {
        const int o  = tid * 16 + p * 4096;  // byte offset, 128B per row
        const int r  = o >> 7;
        const int ch = (o & 127) >> 1;
        async_cp16(Qh + base + (size_t)(mt_blk * 128 + r) * HID + ch,
                   (_Float16*)((char*)Qs + o));
    }

    v8f O[4] = {};
    float mrow[8], lrow[8];
#pragma unroll
    for (int rr = 0; rr < 8; ++rr) { mrow[rr] = -3.0e38f; lrow[rr] = 0.0f; }

    for (int t = 0; t < S / 64; ++t) {
        // ---- stage K tile: TDM descriptor (one wave) or async fallback ----
#if TDM_OK
        if (wave == 0) {
            tdm_load_tile_2d(Kh + base + (size_t)(t * 64) * HID,
                             (unsigned)(size_t)(void*)Ks,
                             /*tile_w=*/64, /*tile_h=*/64, /*pitch=*/HID);
            __builtin_amdgcn_s_wait_tensorcnt(0);
        }
#else
#pragma unroll
        for (int p = 0; p < 2; ++p) {
            const int o  = tid * 16 + p * 4096;
            const int r  = o >> 7;
            const int ch = (o & 127) >> 1;
            async_cp16(Kh + base + (size_t)(t * 64 + r) * HID + ch,
                       (_Float16*)((char*)Ks + o));
        }
#endif
        // ---- stage V transposed (needs shuffle; manual path) ----
        {
            const int r  = tid >> 2;         // 0..63 key row
            const int c0 = (tid & 3) * 16;   // d offset
            const _Float16* vsrc = Vh + base + (size_t)(t * 64 + r) * HID + c0;
            v8h va = *(const v8h*)vsrc;
            v8h vb = *(const v8h*)(vsrc + 8);
#pragma unroll
            for (int i = 0; i < 8; ++i) {
                Vts[(c0 + i) * 64 + r]     = va[i];
                Vts[(c0 + 8 + i) * 64 + r] = vb[i];
            }
            if (t + 1 < S / 64)
                __builtin_prefetch(Vh + base + (size_t)((t + 1) * 64 + r) * HID + c0, 0, 1);
        }
        WAITASYNC(0);      // wave's Q (first iter) / async copies landed
        __syncthreads();

        // ---- S = Q K^T : wave owns 16 query rows x 64 keys ----
        v8f sacc[4] = {};
#pragma unroll
        for (int kc = 0; kc < 2; ++kc) {
            const _Float16* arow = &Qs[(wave * 16 + l16) * 64 + kc * 32];
            v16h a = *(const v16h*)(arow + hl * 16);
#pragma unroll
            for (int nt = 0; nt < 4; ++nt) {
                v16h kb = *(const v16h*)&Ks[(nt * 16 + l16) * 64 + kc * 32 + hl * 16];
                sacc[nt] = wmma_f16(a, kb, sacc[nt]);
            }
        }

        // ---- online softmax per row (16-lane half-wave reductions) ----
#pragma unroll
        for (int rr = 0; rr < 8; ++rr) {
            float mx = sacc[0][rr];
#pragma unroll
            for (int nt = 1; nt < 4; ++nt) mx = fmaxf(mx, sacc[nt][rr]);
#pragma unroll
            for (int msk = 8; msk >= 1; msk >>= 1)
                mx = fmaxf(mx, __shfl_xor(mx, msk, 32));
            const float mnew = fmaxf(mrow[rr], mx);
            const float corr = __expf(mrow[rr] - mnew);
            float psum = 0.0f;
#pragma unroll
            for (int nt = 0; nt < 4; ++nt) {
                const float p = __expf(sacc[nt][rr] - mnew);
                sacc[nt][rr] = p;
                psum += p;
            }
#pragma unroll
            for (int msk = 8; msk >= 1; msk >>= 1)
                psum += __shfl_xor(psum, msk, 32);
            lrow[rr] = lrow[rr] * corr + psum;
            mrow[rr] = mnew;
#pragma unroll
            for (int nt = 0; nt < 4; ++nt) O[nt][rr] *= corr;
        }

        // ---- re-shape P: C-layout regs -> LDS -> A-layout fragments ----
#pragma unroll
        for (int nt = 0; nt < 4; ++nt)
#pragma unroll
            for (int rr = 0; rr < 8; ++rr)
                Ps[(wave * 16 + rr + hl * 8) * 64 + nt * 16 + l16] =
                    (_Float16)sacc[nt][rr];
        __syncthreads();

        // ---- O += P V ----
#pragma unroll
        for (int kc = 0; kc < 2; ++kc) {
            const _Float16* prow = &Ps[(wave * 16 + l16) * 64 + kc * 32];
            v16h pa = *(const v16h*)(prow + hl * 16);
#pragma unroll
            for (int nt = 0; nt < 4; ++nt) {
                v16h vb = *(const v16h*)&Vts[(nt * 16 + l16) * 64 + kc * 32 + hl * 16];
                O[nt] = wmma_f16(pa, vb, O[nt]);
            }
        }
        __syncthreads();
    }

    // ---- normalize and write ctx in [B,S,H*D] layout ----
#pragma unroll
    for (int nt = 0; nt < 4; ++nt)
#pragma unroll
        for (int rr = 0; rr < 8; ++rr) {
            const int s = mt_blk * 128 + wave * 16 + rr + hl * 8;
            const size_t off = base + (size_t)s * HID + nt * 16 + l16;
            Ctx[off] = (_Float16)(O[nt][rr] / lrow[rr]);
        }
}

extern "C" void kernel_launch(void* const* d_in, const int* in_sizes, int n_in,
                              void* d_out, int out_size, void* d_ws, size_t ws_size,
                              hipStream_t stream) {
    const float* query = (const float*)d_in[0];
    const float* key   = (const float*)d_in[1];
    const float* value = (const float*)d_in[2];
    const float* Wq = (const float*)d_in[3];
    const float* bq = (const float*)d_in[4];
    const float* Wk = (const float*)d_in[5];
    const float* bk = (const float*)d_in[6];
    const float* Wv = (const float*)d_in[7];
    const float* bv = (const float*)d_in[8];
    const float* Wo = (const float*)d_in[9];
    const float* bo = (const float*)d_in[10];

    const int M  = in_sizes[0] / HID;           // B*S = 4096
    const int NX = M * HID;                     // input elements
    const int NW = HID * HID;                   // weight elements

    _Float16* p = (_Float16*)d_ws;
    _Float16* Xq  = p; p += NX;
    _Float16* Xk  = p; p += NX;
    _Float16* Xv  = p; p += NX;
    _Float16* Wqh = p; p += NW;
    _Float16* Wkh = p; p += NW;
    _Float16* Wvh = p; p += NW;
    _Float16* Woh = p; p += NW;
    _Float16* Qh  = p; p += NX;
    _Float16* Kh  = p; p += NX;
    _Float16* Vh  = p; p += NX;
    _Float16* Ctx = p; p += NX;

    // f32 -> f16 casts (all sizes are multiples of 2048)
    cast_f16<<<NX / 2048, 256, 0, stream>>>(query, Xq, NX);
    cast_f16<<<NX / 2048, 256, 0, stream>>>(key,   Xk, NX);
    cast_f16<<<NX / 2048, 256, 0, stream>>>(value, Xv, NX);
    cast_f16<<<NW / 2048, 256, 0, stream>>>(Wq, Wqh, NW);
    cast_f16<<<NW / 2048, 256, 0, stream>>>(Wk, Wkh, NW);
    cast_f16<<<NW / 2048, 256, 0, stream>>>(Wv, Wvh, NW);
    cast_f16<<<NW / 2048, 256, 0, stream>>>(Wo, Woh, NW);

    dim3 ggrid(M / BM, HID / BN);   // 32 x 4
    // Q projection with 1/sqrt(head_dim)=1/8 folded in
    gemm_xwT<true><<<ggrid, 256, 0, stream>>>(Xq, Wqh, bq, Qh, M, 0.125f);
    gemm_xwT<true><<<ggrid, 256, 0, stream>>>(Xk, Wkh, bk, Kh, M, 1.0f);
    gemm_xwT<true><<<ggrid, 256, 0, stream>>>(Xv, Wvh, bv, Vh, M, 1.0f);

    // B*H*(S/128) = 2*8*16 = 256 blocks
    flash_attn<<<256, 256, 0, stream>>>(Qh, Kh, Vh, Ctx);

    gemm_xwT<false><<<ggrid, 256, 0, stream>>>(Ctx, Woh, bo, d_out, M, 1.0f);
}
